// GLATransformer_11776800326034
// MI455X (gfx1250) — compile-verified, matrix operand order checked by reading, not run on hardware
//
#include <hip/hip_runtime.h>
#include <hip/hip_bf16.h>

// ---- model constants ----
#define L_   4
#define D_   1024
#define H_   4
#define DK_  128
#define DV_  256
#define KD_  512
#define VD_  1024
#define V_   32000
#define T_   2048
#define LR_  16
#define GLN_ 16.0f
#define EPS_ 1e-5f
#define SCALE_ 0.08838834764831845f   // DK^-0.5

typedef __attribute__((ext_vector_type(16))) __bf16        v16bf;
typedef __attribute__((ext_vector_type(8)))  float         v8f;
typedef __attribute__((ext_vector_type(4)))  unsigned int  v4u;

union Frag16 { v4u q[2]; v16bf v; unsigned short us[16]; };

__device__ __forceinline__ unsigned short f2bf(float f) {
  unsigned int u = __float_as_uint(f);
  u += 0x7FFFu + ((u >> 16) & 1u);      // round-to-nearest-even
  return (unsigned short)(u >> 16);
}

// Fragment-linear packed layout for a [rows x K] bf16 WMMA operand:
// block = (16-row tile, 32-wide K chunk); lane L owns 32 contiguous bytes.
// lane = hi*16 + (row&15); element e -> kk = (j&3)*2 + ((j&4)<<2) + hi*8 + (e&1)
// (lanes 0-15: K {0..7,16..23}, lanes 16-31: K {8..15,24..31}).
__device__ __forceinline__ size_t pack_idx(int row, int k, int K) {
  int chunk = k >> 5, kc = k & 31;
  int hi = (kc >> 3) & 1;
  int kk = kc - hi * 8;                       // {0..7} U {16..23}
  int j  = (kk < 8) ? (kk >> 1) : (4 + ((kk - 16) >> 1));
  int e  = 2 * j + (kk & 1);
  int lane = hi * 16 + (row & 15);
  return ((((size_t)(row >> 4) * (K >> 5) + chunk) * 32) + lane) * 16 + e;
}

// ---------------- weight pack: f32 W[K,N] -> packed bf16 keyed by (n,k) ----------------
__global__ void k_packB(const float* __restrict__ W, unsigned short* __restrict__ dst,
                        int K, int N) {
  int total  = K * N;
  int chunks = K >> 5;
  int i  = blockIdx.x * blockDim.x + threadIdx.x;
  int st = gridDim.x * blockDim.x;
  for (; i < total; i += st) {
    int e     = i & 15;
    int lane  = (i >> 4) & 31;
    int blk   = i >> 9;
    int chunk = blk % chunks;
    int ntile = blk / chunks;
    int hi = lane >> 4;
    int n  = ntile * 16 + (lane & 15);
    int j  = e >> 1;
    int kk = ((j & 3) << 1) + ((j & 4) << 2) + hi * 8 + (e & 1);
    int k  = chunk * 32 + kk;
    dst[i] = f2bf(W[(size_t)k * N + n]);
  }
}

// ---------------- embedding gather: row-major f32 + packed bf16 ----------------
__global__ void k_embed(const int* __restrict__ ids, const float* __restrict__ emb,
                        float* __restrict__ h, unsigned short* __restrict__ hb) {
  int t = blockIdx.x;
  int row = ids[t];
  for (int d = threadIdx.x; d < D_; d += blockDim.x) {
    float x = emb[(size_t)row * D_ + d];
    h[(size_t)t * D_ + d] = x;
    hb[pack_idx(t, d, D_)] = f2bf(x);
  }
}

// ---------------- WMMA GEMM on fragment-packed operands ----------------
// Per wave: 32 rows x 64 cols (2 A frags x 4 B frags -> 8 wmma / K chunk).
// Block = 8 waves arranged 4(M) x 2(N) -> 128 x 128 tile.
// Requires M%128==0, N%128==0, K%32==0.
__global__ __launch_bounds__(256)
void k_gemm_bf16(const unsigned short* __restrict__ A,
                 const unsigned short* __restrict__ B,
                 float* __restrict__ C,
                 unsigned short* __restrict__ Cb,    // packed bf16 shadow (or null)
                 const float* __restrict__ bias,
                 int M, int N, int K, float cscale) {
  const int lane = threadIdx.x & 31;
  const int w    = threadIdx.x >> 5;        // 0..7
  const int wm   = w >> 1;                  // 0..3
  const int wn   = w & 1;                   // 0..1
  const int mt0  = blockIdx.y * 8 + wm * 2; // first of 2 row tiles
  const int nt0  = blockIdx.x * 8 + wn * 4; // first of 4 col tiles
  const int chunks = K >> 5;

  const v4u* Ap0 = (const v4u*)A + ((size_t)(mt0 + 0) * chunks) * 64 + lane * 2;
  const v4u* Ap1 = (const v4u*)A + ((size_t)(mt0 + 1) * chunks) * 64 + lane * 2;
  const v4u* B0p = (const v4u*)B + ((size_t)(nt0 + 0) * chunks) * 64 + lane * 2;
  const v4u* B1p = (const v4u*)B + ((size_t)(nt0 + 1) * chunks) * 64 + lane * 2;
  const v4u* B2p = (const v4u*)B + ((size_t)(nt0 + 2) * chunks) * 64 + lane * 2;
  const v4u* B3p = (const v4u*)B + ((size_t)(nt0 + 3) * chunks) * 64 + lane * 2;

  v8f acc[8];
#pragma unroll
  for (int i = 0; i < 8; ++i) acc[i] = (v8f){};

  for (int c = 0; c < chunks; ++c) {
    Frag16 a0, a1, b0, b1, b2, b3;
    a0.q[0] = Ap0[0]; a0.q[1] = Ap0[1];
    a1.q[0] = Ap1[0]; a1.q[1] = Ap1[1];
    b0.q[0] = B0p[0]; b0.q[1] = B0p[1];
    b1.q[0] = B1p[0]; b1.q[1] = B1p[1];
    b2.q[0] = B2p[0]; b2.q[1] = B2p[1];
    b3.q[0] = B3p[0]; b3.q[1] = B3p[1];
    if (c + 1 < chunks) {                 // global_prefetch_b8 next chunk
      __builtin_prefetch(Ap0 + 64, 0, 1);
      __builtin_prefetch(Ap1 + 64, 0, 1);
      __builtin_prefetch(B0p + 64, 0, 1);
      __builtin_prefetch(B1p + 64, 0, 1);
      __builtin_prefetch(B2p + 64, 0, 1);
      __builtin_prefetch(B3p + 64, 0, 1);
    }
    acc[0] = __builtin_amdgcn_wmma_f32_16x16x32_bf16(false, a0.v, false, b0.v,
                                                     (short)0, acc[0], false, false);
    acc[1] = __builtin_amdgcn_wmma_f32_16x16x32_bf16(false, a0.v, false, b1.v,
                                                     (short)0, acc[1], false, false);
    acc[2] = __builtin_amdgcn_wmma_f32_16x16x32_bf16(false, a0.v, false, b2.v,
                                                     (short)0, acc[2], false, false);
    acc[3] = __builtin_amdgcn_wmma_f32_16x16x32_bf16(false, a0.v, false, b3.v,
                                                     (short)0, acc[3], false, false);
    acc[4] = __builtin_amdgcn_wmma_f32_16x16x32_bf16(false, a1.v, false, b0.v,
                                                     (short)0, acc[4], false, false);
    acc[5] = __builtin_amdgcn_wmma_f32_16x16x32_bf16(false, a1.v, false, b1.v,
                                                     (short)0, acc[5], false, false);
    acc[6] = __builtin_amdgcn_wmma_f32_16x16x32_bf16(false, a1.v, false, b2.v,
                                                     (short)0, acc[6], false, false);
    acc[7] = __builtin_amdgcn_wmma_f32_16x16x32_bf16(false, a1.v, false, b3.v,
                                                     (short)0, acc[7], false, false);
    Ap0 += 64; Ap1 += 64; B0p += 64; B1p += 64; B2p += 64; B3p += 64;
  }

  const int hi = lane >> 4;
  const int lr = lane & 15;
#pragma unroll
  for (int ai = 0; ai < 2; ++ai) {
#pragma unroll
    for (int ti = 0; ti < 4; ++ti) {
      int col = (nt0 + ti) * 16 + lr;
      float bsv = bias ? bias[col] : 0.f;
#pragma unroll
      for (int j = 0; j < 8; ++j) {
        int row = (mt0 + ai) * 16 + j + 8 * hi;
        float x = acc[ai * 4 + ti][j] * cscale + bsv;
        C[(size_t)row * N + col] = x;
        if (Cb) Cb[pack_idx(row, col, N)] = f2bf(x);
      }
    }
  }
}

// ---------------- low-rank forget-gate path (f32, tiny) ----------------
__global__ __launch_bounds__(256)
void k_lr1(const float* __restrict__ h, const float* __restrict__ W1,
           float* __restrict__ out) {
  int t = blockIdx.x;
  int col = threadIdx.x & 15;
  int seg = threadIdx.x >> 4;
  __shared__ float red[256];
  float acc = 0.f;
  int kb = seg * 64;
  for (int k = kb; k < kb + 64; ++k)
    acc += h[(size_t)t * D_ + k] * W1[(size_t)k * LR_ + col];
  red[threadIdx.x] = acc;
  __syncthreads();
  for (int s = 8; s > 0; s >>= 1) {
    if (seg < s) red[seg * 16 + col] += red[(seg + s) * 16 + col];
    __syncthreads();
  }
  if (seg == 0) out[(size_t)t * LR_ + col] = red[col];
}

__global__ __launch_bounds__(512)
void k_lr2(const float* __restrict__ xg, const float* __restrict__ W2,
           const float* __restrict__ bg, float* __restrict__ gk) {
  int t = blockIdx.x, n = threadIdx.x;
  float acc = bg[n];
#pragma unroll
  for (int r = 0; r < LR_; ++r) acc += xg[(size_t)t * LR_ + r] * W2[r * KD_ + n];
  float ls = (acc >= 0.f) ? -log1pf(__expf(-acc)) : (acc - log1pf(__expf(acc)));
  gk[(size_t)t * KD_ + n] = ls / GLN_;
}

// ---------------- sequential GLA scan: one block per head ----------------
__global__ __launch_bounds__(512)
void k_scan(const float* __restrict__ q, const float* __restrict__ k,
            const float* __restrict__ v, const float* __restrict__ gk,
            float* __restrict__ o) {
  const int h    = blockIdx.x;
  const int tid  = threadIdx.x;
  const int vcol = tid >> 1;
  const int half = tid & 1;
  const int kb   = half * 64;
  __shared__ float q_s[DK_], k_s[DK_], e_s[DK_];

  float S[64];
#pragma unroll
  for (int j = 0; j < 64; ++j) S[j] = 0.f;

  for (int t = 0; t < T_; ++t) {
    __syncthreads();
    if (tid < 128)      q_s[tid]       = q[(size_t)t * KD_ + h * DK_ + tid];
    else if (tid < 256) k_s[tid - 128] = k[(size_t)t * KD_ + h * DK_ + (tid - 128)];
    else if (tid < 384) e_s[tid - 256] = __expf(gk[(size_t)t * KD_ + h * DK_ + (tid - 256)]);
    __syncthreads();
    float vv = v[(size_t)t * VD_ + h * DV_ + vcol];
    float acc = 0.f;
#pragma unroll
    for (int j = 0; j < 64; ++j) {
      float s = S[j] * e_s[kb + j] + k_s[kb + j] * vv;
      S[j] = s;
      acc += q_s[kb + j] * s;
    }
    acc += __shfl_xor(acc, 1, 32);
    if (half == 0) o[(size_t)t * VD_ + h * DV_ + vcol] = acc;
  }
}

// ------------- fused RMSNorm (per head over DV) + swish gate; packed bf16 out -------------
__global__ __launch_bounds__(256)
void k_rms_gate(const float* __restrict__ o, const float* __restrict__ g,
                const float* __restrict__ nw, unsigned short* __restrict__ ob) {
  int t = blockIdx.x >> 2;     // H_ == 4
  int h = blockIdx.x & 3;
  int dv = threadIdx.x;
  __shared__ float red[256];
  size_t idx = (size_t)t * VD_ + h * DV_ + dv;
  float x = o[idx];
  red[dv] = x * x;
  __syncthreads();
  for (int s = 128; s > 0; s >>= 1) {
    if (dv < s) red[dv] += red[dv + s];
    __syncthreads();
  }
  float r  = rsqrtf(red[0] / (float)DV_ + EPS_);
  float gv = g[idx];
  float sw = gv / (1.f + __expf(-gv));
  ob[pack_idx(t, h * DV_ + dv, VD_)] = f2bf(x * r * nw[dv] * sw);
}

// ---------------- final LayerNorm over D, packed bf16 out ----------------
__global__ __launch_bounds__(256)
void k_ln(const float* __restrict__ h, const float* __restrict__ w,
          const float* __restrict__ b, unsigned short* __restrict__ hb) {
  int t = blockIdx.x, tid = threadIdx.x;
  __shared__ float rs[256], rq[256];
  float s = 0.f, q = 0.f, xv[4];
#pragma unroll
  for (int i = 0; i < 4; ++i) {
    float x = h[(size_t)t * D_ + tid + i * 256];
    xv[i] = x; s += x; q += x * x;
  }
  rs[tid] = s; rq[tid] = q;
  __syncthreads();
  for (int st = 128; st > 0; st >>= 1) {
    if (tid < st) { rs[tid] += rs[tid + st]; rq[tid] += rq[tid + st]; }
    __syncthreads();
  }
  float mu  = rs[0] / (float)D_;
  float var = rq[0] / (float)D_ - mu * mu;
  float inv = rsqrtf(var + EPS_);
#pragma unroll
  for (int i = 0; i < 4; ++i) {
    int d = tid + i * 256;
    hb[pack_idx(t, d, D_)] = f2bf((xv[i] - mu) * inv * w[d] + b[d]);
  }
}

// ---------------- host orchestration ----------------
static void gemm(const unsigned short* A, const unsigned short* B, float* C,
                 unsigned short* Cb, const float* bias, int M, int N, int K,
                 float cscale, hipStream_t s) {
  dim3 g(N / 128, M / 128);
  k_gemm_bf16<<<g, 256, 0, s>>>(A, B, C, Cb, bias, M, N, K, cscale);
}

static void packB(const float* W, unsigned short* dst, int K, int N, hipStream_t st) {
  int n = K * N;
  int blocks = (n + 255) / 256;
  if (blocks > 4096) blocks = 4096;
  k_packB<<<blocks, 256, 0, st>>>(W, dst, K, N);
}

extern "C" void kernel_launch(void* const* d_in, const int* in_sizes, int n_in,
                              void* d_out, int out_size, void* d_ws, size_t ws_size,
                              hipStream_t stream) {
  (void)in_sizes; (void)n_in; (void)out_size; (void)ws_size;

  const int*   ids  = (const int*)  d_in[0];
  const float* emb  = (const float*)d_in[1];
  const float* Wq   = (const float*)d_in[2];
  const float* Wk   = (const float*)d_in[3];
  const float* Wv   = (const float*)d_in[4];
  const float* Wg   = (const float*)d_in[5];
  const float* Wgk1 = (const float*)d_in[6];
  const float* Wgk2 = (const float*)d_in[7];
  const float* bgk  = (const float*)d_in[8];
  const float* gnw  = (const float*)d_in[9];
  const float* Wo   = (const float*)d_in[10];
  const float* lnw  = (const float*)d_in[11];
  const float* lnb  = (const float*)d_in[12];
  const float* Wlm  = (const float*)d_in[13];
  const float* blm  = (const float*)d_in[14];
  float* out = (float*)d_out;

  char* p = (char*)d_ws;
  auto carve = [&](size_t bytes) -> void* {
    void* r = (void*)p;
    p += (bytes + 255) & ~(size_t)255;
    return r;
  };

  unsigned short* wq_b  = (unsigned short*)carve((size_t)L_ * D_ * KD_ * 2);
  unsigned short* wk_b  = (unsigned short*)carve((size_t)L_ * D_ * KD_ * 2);
  unsigned short* wv_b  = (unsigned short*)carve((size_t)L_ * D_ * VD_ * 2);
  unsigned short* wg_b  = (unsigned short*)carve((size_t)L_ * D_ * VD_ * 2);
  unsigned short* wo_b  = (unsigned short*)carve((size_t)L_ * VD_ * D_ * 2);
  unsigned short* wlm_b = (unsigned short*)carve((size_t)D_ * V_ * 2);
  float*          hf    = (float*)carve((size_t)T_ * D_ * 4);
  unsigned short* hb    = (unsigned short*)carve((size_t)T_ * D_ * 2);
  float*          qbuf  = (float*)carve((size_t)T_ * KD_ * 4);
  float*          kbuf  = (float*)carve((size_t)T_ * KD_ * 4);
  float*          vbuf  = (float*)carve((size_t)T_ * VD_ * 4);
  float*          gbuf  = (float*)carve((size_t)T_ * VD_ * 4);
  float*          gkbuf = (float*)carve((size_t)T_ * KD_ * 4);
  float*          xg1   = (float*)carve((size_t)T_ * LR_ * 4);
  float*          obuf  = (float*)carve((size_t)T_ * VD_ * 4);
  unsigned short* ob    = (unsigned short*)carve((size_t)T_ * VD_ * 2);
  unsigned short* hnb   = (unsigned short*)carve((size_t)T_ * D_ * 2);

  // per-launch weight packing into WMMA fragment-linear bf16 layout
  for (int i = 0; i < L_; ++i) {
    packB(Wq + (size_t)i * D_ * KD_, wq_b + (size_t)i * D_ * KD_, D_, KD_, stream);
    packB(Wk + (size_t)i * D_ * KD_, wk_b + (size_t)i * D_ * KD_, D_, KD_, stream);
    packB(Wv + (size_t)i * D_ * VD_, wv_b + (size_t)i * D_ * VD_, D_, VD_, stream);
    packB(Wg + (size_t)i * D_ * VD_, wg_b + (size_t)i * D_ * VD_, D_, VD_, stream);
    packB(Wo + (size_t)i * VD_ * D_, wo_b + (size_t)i * VD_ * D_, VD_, D_, stream);
  }
  packB(Wlm, wlm_b, D_, V_, stream);

  k_embed<<<T_, 256, 0, stream>>>(ids, emb, hf, hb);

  for (int i = 0; i < L_; ++i) {
    size_t oqk = (size_t)i * D_ * KD_;
    size_t ovg = (size_t)i * D_ * VD_;
    gemm(hb, wq_b + oqk, qbuf, nullptr, nullptr, T_, KD_, D_, SCALE_, stream);
    gemm(hb, wk_b + oqk, kbuf, nullptr, nullptr, T_, KD_, D_, 1.f, stream);
    gemm(hb, wv_b + ovg, vbuf, nullptr, nullptr, T_, VD_, D_, 1.f, stream);
    gemm(hb, wg_b + ovg, gbuf, nullptr, nullptr, T_, VD_, D_, 1.f, stream);
    k_lr1<<<T_, 256, 0, stream>>>(hf, Wgk1 + (size_t)i * D_ * LR_, xg1);
    k_lr2<<<T_, 512, 0, stream>>>(xg1, Wgk2 + (size_t)i * LR_ * KD_,
                                  bgk + (size_t)i * KD_, gkbuf);
    k_scan<<<H_, 512, 0, stream>>>(qbuf, kbuf, vbuf, gkbuf, obuf);
    k_rms_gate<<<T_ * H_, 256, 0, stream>>>(obuf, gbuf, gnw + (size_t)i * DV_, ob);
    gemm(ob, wo_b + (size_t)i * VD_ * D_, hf, hb, nullptr, T_, D_, VD_, 1.f, stream);
  }

  k_ln<<<T_, 256, 0, stream>>>(hf, lnw, lnb, hnb);
  gemm(hnb, wlm_b, out, nullptr, blm, T_, V_, D_, 1.f, stream);
}
